// FlattenedVectorQuantizer_76897094468432
// MI455X (gfx1250) — matrix operand order repaired and checked by reading, not compiled, not run on hardware
//
#include <hip/hip_runtime.h>
#include <hip/hip_bf16.h>
#include <stdint.h>

// Problem constants (match reference).
#define N_TOK   65536
#define K_CODES 4096
#define DIM     256
#define NTILES  (K_CODES / 16)
#define TILE_DW 2048   // 16 codes x 256 bf16 / 2 per dword

typedef __attribute__((ext_vector_type(16))) __bf16 v16bf;
typedef __attribute__((ext_vector_type(8)))  float  v8f;

union Frag { uint32_t u[8]; v16bf v; };   // 16 bf16 = 8 dwords
union AccU { float f[8];    v8f  v; };    // 16x16 f32 C/D tile slice

__device__ __forceinline__ uint32_t f2u(float x)    { union { float f; uint32_t u; } c; c.f = x; return c.u; }
__device__ __forceinline__ float    u2f(uint32_t u) { union { float f; uint32_t u; } c; c.u = u; return c.f; }

// 8 consecutive f32 -> 4 packed-bf16 dwords hi (truncate) + lo (RNE residual).
__device__ __forceinline__ void cvt8(const float* x, uint32_t* hi, uint32_t* lo) {
#pragma unroll
  for (int j = 0; j < 4; ++j) {
    float a = x[2*j], b = x[2*j + 1];
    uint32_t ha = f2u(a) & 0xFFFF0000u;
    uint32_t hb = f2u(b) & 0xFFFF0000u;
    float la = a - u2f(ha);
    float lb = b - u2f(hb);
    uint32_t ra = f2u(la); ra = (ra + 0x7FFFu + ((ra >> 16) & 1u)) & 0xFFFF0000u;
    uint32_t rb = f2u(lb); rb = (rb + 0x7FFFu + ((rb >> 16) & 1u)) & 0xFFFF0000u;
    hi[j] = (ha >> 16) | hb;
    lo[j] = (ra >> 16) | rb;
  }
}

// CDNA5 async global->LDS staging (ASYNCcnt-tracked), with sync fallback.
#if defined(__gfx1250__) && __has_builtin(__builtin_amdgcn_global_load_async_to_lds_b128)
#define VQ_ASYNC 1
#endif

#ifdef VQ_ASYNC
// Parameter type per hipcc diagnostic: 'int __vector(4) __device__ *'
// (addrspace(1) prints as __device__ in HIP mode; LDS side is addrspace(3)).
typedef int vq_i4 __attribute__((__vector_size__(4 * sizeof(int))));
typedef __attribute__((address_space(1))) vq_i4 vq_gi4;
typedef __attribute__((address_space(3))) vq_i4 vq_li4;
#endif

__device__ __forceinline__ void vq_wait_async() {
#ifdef VQ_ASYNC
#if __has_builtin(__builtin_amdgcn_s_wait_asynccnt)
  __builtin_amdgcn_s_wait_asynccnt(0);
#else
  asm volatile("s_wait_asynccnt 0x0" ::: "memory");
#endif
#endif
}

// Copy one pre-converted 16-code tile (2048 dwords hi + 2048 dwords lo,
// both contiguous in global) into LDS. 256 threads, 4 x b128 chunks each.
__device__ __forceinline__ void vq_copy_tile(const uint32_t* __restrict__ gh,
                                             const uint32_t* __restrict__ gl,
                                             uint32_t* lh, uint32_t* ll, int tid) {
#ifdef VQ_ASYNC
  __builtin_amdgcn_global_load_async_to_lds_b128(
      (vq_gi4*)(gh + tid * 4),        (vq_li4*)(lh + tid * 4),        0, 0);
  __builtin_amdgcn_global_load_async_to_lds_b128(
      (vq_gi4*)(gh + 1024 + tid * 4), (vq_li4*)(lh + 1024 + tid * 4), 0, 0);
  __builtin_amdgcn_global_load_async_to_lds_b128(
      (vq_gi4*)(gl + tid * 4),        (vq_li4*)(ll + tid * 4),        0, 0);
  __builtin_amdgcn_global_load_async_to_lds_b128(
      (vq_gi4*)(gl + 1024 + tid * 4), (vq_li4*)(ll + 1024 + tid * 4), 0, 0);
#else
  uint4 a = *(const uint4*)(gh + tid * 4);
  uint4 b = *(const uint4*)(gh + 1024 + tid * 4);
  uint4 c = *(const uint4*)(gl + tid * 4);
  uint4 d = *(const uint4*)(gl + 1024 + tid * 4);
  *(uint4*)(lh + tid * 4)        = a;
  *(uint4*)(lh + 1024 + tid * 4) = b;
  *(uint4*)(ll + tid * 4)        = c;
  *(uint4*)(ll + 1024 + tid * 4) = d;
#endif
}

// ---------------------------------------------------------------------------
// Kernel A: one-shot codebook f32 -> packed bf16 hi/lo (done once, not 512x).
// ---------------------------------------------------------------------------
__global__ __launch_bounds__(256) void vq_cvt_cb_kernel(const float* __restrict__ cb,
                                                        uint32_t* __restrict__ cbh,
                                                        uint32_t* __restrict__ cbl) {
  const size_t base = ((size_t)blockIdx.x * 256 + threadIdx.x) * 8;  // 8 f32 / thread
  float v[8];
  *(float4*)&v[0] = *(const float4*)(cb + base);
  *(float4*)&v[4] = *(const float4*)(cb + base + 4);
  uint32_t h[4], l[4];
  cvt8(v, h, l);
  *(uint4*)(cbh + base / 2) = make_uint4(h[0], h[1], h[2], h[3]);
  *(uint4*)(cbl + base / 2) = make_uint4(l[0], l[1], l[2], l[3]);
}

// ---------------------------------------------------------------------------
// Kernel B: e_sq[k] = ||codebook[k]||^2, one wave per code.
// ---------------------------------------------------------------------------
__global__ __launch_bounds__(256) void vq_esq_kernel(const float* __restrict__ cb,
                                                     float* __restrict__ esq) {
  int wave = (blockIdx.x * blockDim.x + threadIdx.x) >> 5;
  int lane = threadIdx.x & 31;
  if (wave >= K_CODES) return;
  const float* p = cb + (size_t)wave * DIM;
  float s = 0.0f;
#pragma unroll
  for (int i = 0; i < DIM / 32; ++i) { float v = p[lane + 32 * i]; s += v * v; }
#pragma unroll
  for (int m = 16; m >= 1; m >>= 1) s += __shfl_xor(s, m, 32);
  if (lane == 0) esq[wave] = s;
}

// ---------------------------------------------------------------------------
// Kernel C: fused (-2 z.e + ||e||^2) + argmin via bf16 split WMMA.
// 256 threads = 8 waves = 128 rows/block; double-buffered async LDS staging.
// ---------------------------------------------------------------------------
__global__ __launch_bounds__(256) void vq_argmin_kernel(const float* __restrict__ z,
                                                        const uint32_t* __restrict__ cbh,
                                                        const uint32_t* __restrict__ cbl,
                                                        const float* __restrict__ esq,
                                                        float* __restrict__ idx_out_f) {
  __shared__ uint32_t bhi[2 * TILE_DW];  // double-buffered hi tiles (16 KB)
  __shared__ uint32_t blo[2 * TILE_DW];  // double-buffered lo tiles (16 KB)

  const int tid  = threadIdx.x;
  const int lane = tid & 31;
  const int half = lane >> 4;    // K-half selector per ISA fragment layout
  const int r16  = lane & 15;    // A row / B column within tile
  const int wave = tid >> 5;
  const int rowBase = blockIdx.x * 128 + wave * 16;

  // --- Load this wave's 16 z-rows once; keep hi/lo A fragments in VGPRs. ---
  // A 16-bit 16x32 layout: VGPR j<4 -> K = 2j + 8*half;  j>=4 -> K = 16+2(j-4)+8*half.
  Frag a_hi[8], a_lo[8];
  {
    const float* zr = z + (size_t)(rowBase + r16) * DIM;
#pragma unroll
    for (int s = 0; s < 8; ++s) {
      const int kb = s * 32 + half * 8;
      float xs[8], ys[8];
      *(float4*)&xs[0] = *(const float4*)(zr + kb);
      *(float4*)&xs[4] = *(const float4*)(zr + kb + 4);
      *(float4*)&ys[0] = *(const float4*)(zr + kb + 16);
      *(float4*)&ys[4] = *(const float4*)(zr + kb + 20);
      cvt8(xs, &a_hi[s].u[0], &a_lo[s].u[0]);
      cvt8(ys, &a_hi[s].u[4], &a_lo[s].u[4]);
    }
  }

  float best[8];
  int   bidx[8];
#pragma unroll
  for (int r = 0; r < 8; ++r) { best[r] = 3.4e38f; bidx[r] = 0; }

  // Prologue: stage tile 0 into buffer 0.
  vq_copy_tile(cbh, cbl, &bhi[0], &blo[0], tid);

  for (int t = 0; t < NTILES; ++t) {
    const int buf = (t & 1) ? TILE_DW : 0;
    vq_wait_async();      // tile t's async copies landed in LDS (no-op if sync path)
    __syncthreads();      // visible to all waves

    if (t + 1 < NTILES)   // overlap next tile's copy with this tile's WMMAs
      vq_copy_tile(cbh + (size_t)(t + 1) * TILE_DW, cbl + (size_t)(t + 1) * TILE_DW,
                   &bhi[buf ^ TILE_DW], &blo[buf ^ TILE_DW], tid);

    AccU acc;
#pragma unroll
    for (int r = 0; r < 8; ++r) acc.f[r] = 0.0f;

    // D = 256 in 8 k-steps; 3-term bf16 split: hi*hi + lo*hi + hi*lo.
#pragma unroll
    for (int s = 0; s < 8; ++s) {
      // B 16-bit 32x16 layout (matches A): VGPR j<4 -> K = 2j + 8*half;
      // j>=4 -> K = 16+2(j-4)+8*half -> two 4-dword LDS chunks per fragment.
      const uint32_t* ph = &bhi[buf + r16 * 128 + s * 16 + half * 4];
      const uint32_t* pl = &blo[buf + r16 * 128 + s * 16 + half * 4];
      Frag b_h, b_l;
      ((uint4*)b_h.u)[0] = ((const uint4*)ph)[0];
      ((uint4*)b_h.u)[1] = ((const uint4*)ph)[2];
      ((uint4*)b_l.u)[0] = ((const uint4*)pl)[0];
      ((uint4*)b_l.u)[1] = ((const uint4*)pl)[2];
      acc.v = __builtin_amdgcn_wmma_f32_16x16x32_bf16(false, a_hi[s].v, false, b_h.v,
                                                      (short)0, acc.v, false, false);
      acc.v = __builtin_amdgcn_wmma_f32_16x16x32_bf16(false, a_lo[s].v, false, b_h.v,
                                                      (short)0, acc.v, false, false);
      acc.v = __builtin_amdgcn_wmma_f32_16x16x32_bf16(false, a_hi[s].v, false, b_l.v,
                                                      (short)0, acc.v, false, false);
    }

    // C/D layout: lane holds column n=r16; VGPR r holds row 8*half + r.
    const float es  = esq[t * 16 + r16];
    const int  code = t * 16 + r16;
#pragma unroll
    for (int r = 0; r < 8; ++r) {
      float sc = es - 2.0f * acc.f[r];
      if (sc < best[r]) { best[r] = sc; bidx[r] = code; }
    }
  }

  // Argmin across the 16 columns (lanes within each half), first-index tie-break.
#pragma unroll
  for (int r = 0; r < 8; ++r) {
    float v = best[r];
    int   i = bidx[r];
#pragma unroll
    for (int m = 1; m < 16; m <<= 1) {
      float ov = __shfl_xor(v, m, 16);
      int   oi = __shfl_xor(i, m, 16);
      if (ov < v || (ov == v && oi < i)) { v = ov; i = oi; }
    }
    best[r] = v; bidx[r] = i;
  }
  if (r16 == 0) {
#pragma unroll
    for (int r = 0; r < 8; ++r)
      idx_out_f[rowBase + half * 8 + r] = (float)bidx[r];
  }
}

// ---------------------------------------------------------------------------
// Kernel D: init loss accumulator.
// ---------------------------------------------------------------------------
__global__ void vq_init_kernel(float* acc) { *acc = 0.0f; }

// ---------------------------------------------------------------------------
// Kernel E: gather quantized rows (codebook is L2-resident) + loss partial sum.
// ---------------------------------------------------------------------------
__global__ __launch_bounds__(256) void vq_gather_loss_kernel(const float* __restrict__ z,
                                                             const float* __restrict__ cb,
                                                             const float* __restrict__ idx_f,
                                                             float* __restrict__ q_out,
                                                             float* __restrict__ acc) {
  __shared__ float red[8];
  const size_t g    = (size_t)blockIdx.x * 256 + threadIdx.x;
  const int    row  = (int)(g >> 8);
  const int    d    = (int)(g & 255);
  const int    code = (int)idx_f[row];
  const float  q  = cb[(size_t)code * DIM + d];
  const float  zz = z[g];
  q_out[g] = q;  // quantized_st == quantized numerically
  float diff = q - zz;
  float s = diff * diff;
#pragma unroll
  for (int m = 16; m >= 1; m >>= 1) s += __shfl_xor(s, m, 32);
  const int lane = threadIdx.x & 31, w = threadIdx.x >> 5;
  if (lane == 0) red[w] = s;
  __syncthreads();
  if (threadIdx.x == 0) {
    float t = 0.0f;
#pragma unroll
    for (int i = 0; i < 8; ++i) t += red[i];
    atomicAdd(acc, t);
  }
}

// ---------------------------------------------------------------------------
// Kernel F: loss = 1.25 * sum / (N*D)  (q_latent + 0.25*e_latent, equal means)
// ---------------------------------------------------------------------------
__global__ void vq_finalize_kernel(const float* acc, float* loss) {
  *loss = *acc * (1.25f / ((float)N_TOK * (float)DIM));
}

extern "C" void kernel_launch(void* const* d_in, const int* in_sizes, int n_in,
                              void* d_out, int out_size, void* d_ws, size_t ws_size,
                              hipStream_t stream) {
  const float* z  = (const float*)d_in[0];   // [N, D] f32
  const float* cb = (const float*)d_in[1];   // [K, D] f32

  float* out  = (float*)d_out;
  float* loss = out;                                   // [1]
  float* qout = out + 1;                               // [N*D]
  float* idxf = out + 1 + (size_t)N_TOK * DIM;         // [N] (indices as f32)

  // Scratch: converted bf16 codebook lives inside the quantized output region
  // (overwritten by the gather kernel afterwards). 16B-aligned at out+16.
  uint32_t* cbh = (uint32_t*)(out + 16);               // K*D/2 dwords (2 MB)
  uint32_t* cbl = cbh + (size_t)K_CODES * DIM / 2;     // K*D/2 dwords (2 MB)

  float* ws  = (float*)d_ws;
  float* acc = ws;        // 1 float
  float* esq = ws + 64;   // K floats

  hipLaunchKernelGGL(vq_init_kernel, dim3(1), dim3(1), 0, stream, acc);
  hipLaunchKernelGGL(vq_cvt_cb_kernel, dim3(K_CODES * DIM / 2048), dim3(256), 0, stream,
                     cb, cbh, cbl);
  hipLaunchKernelGGL(vq_esq_kernel, dim3(K_CODES / 8), dim3(256), 0, stream, cb, esq);
  hipLaunchKernelGGL(vq_argmin_kernel, dim3(N_TOK / 128), dim3(256), 0, stream,
                     z, cbh, cbl, esq, idxf);
  hipLaunchKernelGGL(vq_gather_loss_kernel, dim3((N_TOK * DIM) / 256), dim3(256), 0, stream,
                     z, cb, idxf, qout, acc);
  hipLaunchKernelGGL(vq_finalize_kernel, dim3(1), dim3(1), 0, stream, acc, loss);
}
